// SupConLoss_28363964023058
// MI455X (gfx1250) — matrix operand order repaired and checked by reading, not compiled, not run on hardware
//
#include <hip/hip_runtime.h>
#include <hip/hip_bf16.h>

// ---------------------------------------------------------------------------
// SupConLoss fused kernel for MI455X (gfx1250, wave32, WMMA).
//
// B=6144 anchors, NCOL=2B=12288 contrast rows, D=128, T=0.07.
// s[i,j] = dot(anchor_i, contrast_j)/T computed tile-by-tile with
// 3x V_WMMA_F32_16X16X32_BF16 (split bf16 hi/lo, f32 accumulate) and fused
// directly into per-row softmax statistics; the 302MB s matrix is never
// materialized. Since features are unit vectors, |s| <= 1/T, so exp uses the
// fixed stabilizer C = 1/T (no online rescaling) and the reference's exact
// max_all / max_opp terms are reconstructed analytically at finalize.
//
// 8 waves per block (2+ waves/SIMD32) so L1/L2 load latency is hidden by
// wave-level parallelism in addition to the compiler's software pipelining;
// WMMA is TRANS-tracked on CDNA5 and co-executes with the softmax VALU work.
// ---------------------------------------------------------------------------

#define BROWS 6144
#define DIM   128
#define NCOL  12288
#define TEMP  0.07f
#define NEGBIG (-3.0e38f)
#define NWAVES 8

typedef __attribute__((ext_vector_type(16))) __bf16 v16bf;
typedef __attribute__((ext_vector_type(8)))  __bf16 bf8;
typedef __attribute__((ext_vector_type(8)))  float  v8f;

// ---- f32 -> bf16 (RNE) via integer ops (avoids relying on __bf16 arithmetic)
static __device__ inline unsigned short f32_to_bf16_rne(float x) {
  unsigned u = __float_as_uint(x);
  unsigned r = u + 0x7FFFu + ((u >> 16) & 1u);
  return (unsigned short)(r >> 16);
}

// Load 16 bf16 elements (two aligned 16B chunks) into a v16bf WMMA operand.
static __device__ inline v16bf ld16(const unsigned short* p0, const unsigned short* p1) {
  bf8 a = *reinterpret_cast<const bf8*>(p0);
  bf8 b = *reinterpret_cast<const bf8*>(p1);
  return __builtin_shufflevector(a, b, 0,1,2,3,4,5,6,7,8,9,10,11,12,13,14,15);
}

// ---------------------------------------------------------------------------
// Kernel 1: convert features (B,V,D) f32 into contrast-ordered bf16 hi/lo
// arrays: contrast[j][d] = features[j%B][j/B][d]. hi = RNE(x), lo = RNE(x-hi).
// ---------------------------------------------------------------------------
__global__ void supcon_prep(const float* __restrict__ feat,
                            unsigned short* __restrict__ hi,
                            unsigned short* __restrict__ lo) {
  int idx = blockIdx.x * blockDim.x + threadIdx.x;      // 0 .. NCOL*DIM-1
  if (idx >= NCOL * DIM) return;
  int j = idx >> 7;
  int d = idx & 127;
  int b = (j < BROWS) ? j : (j - BROWS);
  int v = (j < BROWS) ? 0 : 1;
  float x = feat[((size_t)b * 2 + v) * DIM + d];
  unsigned short h = f32_to_bf16_rne(x);
  float hf = __uint_as_float(((unsigned)h) << 16);
  hi[idx] = h;
  lo[idx] = f32_to_bf16_rne(x - hf);
}

// ---------------------------------------------------------------------------
// Kernel 2: fused GEMM + row statistics. One block = 16 anchor rows.
// 8 waves split the 768 column tiles (interleaved for L2 locality).
// Per 16x16 tile: 12 bf16 WMMAs, then per-element online updates:
//   e = exp(s - C); running max_all, sum_all (excl. diag), sum/count over
//   positives, max/sum over opposite-label columns, and s[i, i+B].
// ---------------------------------------------------------------------------
__launch_bounds__(NWAVES * 32)
__global__ void supcon_main(const unsigned short* __restrict__ hi,
                            const unsigned short* __restrict__ lo,
                            const int*   __restrict__ labels,
                            const float* __restrict__ weight,
                            float* __restrict__ rowloss,
                            float* __restrict__ rowelim) {
  const float invT = 1.0f / TEMP;
  const float CMAX = 1.0f / TEMP;       // |s| <= 1/T for unit features
  __shared__ float comb[NWAVES][16][8];

  const int tid  = threadIdx.x;
  const int w    = tid >> 5;
  const int lane = tid & 31;
  const int half = lane >> 4;
  const int ln   = lane & 15;
  const int row_base = blockIdx.x * 16;

  // --- A operand (16 rows x 128): lane ln holds row M=ln;
  //     lower half-wave K=0..15+32*kt, upper half K=16..31+32*kt.
  const unsigned short* arow_hi = hi + (size_t)(row_base + ln) * DIM;
  const unsigned short* arow_lo = lo + (size_t)(row_base + ln) * DIM;
  const int k0a = half * 16;
  v16bf a_hi[4], a_lo[4];
#pragma unroll
  for (int kt = 0; kt < 4; ++kt) {
    int d0 = kt * 32 + k0a;
    a_hi[kt] = ld16(arow_hi + d0, arow_hi + d0 + 8);
    a_lo[kt] = ld16(arow_lo + d0, arow_lo + d0 + 8);
  }

  // Row labels for the 8 C/D rows this lane owns (row = r + 8*half).
  int labrow[8];
#pragma unroll
  for (int r = 0; r < 8; ++r) labrow[r] = labels[row_base + r + 8 * half];

  float mx_all[8], sum_all[8], spos[8], cpos[8], mopp[8], sopp[8], firsts[8];
#pragma unroll
  for (int r = 0; r < 8; ++r) {
    mx_all[r] = NEGBIG; sum_all[r] = 0.0f; spos[r] = 0.0f; cpos[r] = 0.0f;
    mopp[r]   = NEGBIG; sopp[r]   = 0.0f; firsts[r] = NEGBIG;
  }

  // --- B operand layout: lane ln holds column N=ln; lower half K=0..15,
  //     upper half K=16..31 (consecutive within lane), per 32-K tile.
  const int k0b = half * 16;
  for (int ct = w; ct < NCOL / 16; ct += NWAVES) {
    const int j = ct * 16 + ln;
    const unsigned short* brow_hi = hi + (size_t)j * DIM;
    const unsigned short* brow_lo = lo + (size_t)j * DIM;

    v8f acc = {};
#pragma unroll
    for (int kt = 0; kt < 4; ++kt) {
      int d0 = kt * 32 + k0b;
      v16bf b_hi = ld16(brow_hi + d0, brow_hi + d0 + 8);
      v16bf b_lo = ld16(brow_lo + d0, brow_lo + d0 + 8);
      // split-bf16 fp32 emulation: hi*hi + hi*lo + lo*hi (f32 accumulate)
      acc = __builtin_amdgcn_wmma_f32_16x16x32_bf16(false, a_hi[kt], false, b_hi,
                                                    (short)0, acc, false, false);
      acc = __builtin_amdgcn_wmma_f32_16x16x32_bf16(false, a_hi[kt], false, b_lo,
                                                    (short)0, acc, false, false);
      acc = __builtin_amdgcn_wmma_f32_16x16x32_bf16(false, a_lo[kt], false, b_hi,
                                                    (short)0, acc, false, false);
    }

    const int labc = labels[(j < BROWS) ? j : (j - BROWS)];
#pragma unroll
    for (int r = 0; r < 8; ++r) {
      const int i = row_base + r + 8 * half;   // C/D row of VGPR r
      float s = acc[r] * invT;
      float e = __expf(s - CMAX);              // always finite, <= 1
      bool diag = (j == i);
      bool pos  = (labc == labrow[r]);
      mx_all[r] = fmaxf(mx_all[r], s);
      sum_all[r] += diag ? 0.0f : e;
      float take = (pos && !diag) ? 1.0f : 0.0f;
      spos[r] += take * s;
      cpos[r] += take;
      sopp[r] += pos ? 0.0f : e;
      mopp[r]  = fmaxf(mopp[r], pos ? NEGBIG : s);
      firsts[r] = (j == i + BROWS) ? s : firsts[r];
    }
  }

  // --- reduce across the 16 lanes sharing each C/D row (xor stays in-half)
#pragma unroll
  for (int r = 0; r < 8; ++r) {
#pragma unroll
    for (int o = 1; o < 16; o <<= 1) {
      mx_all[r]  = fmaxf(mx_all[r], __shfl_xor(mx_all[r], o, 32));
      sum_all[r] += __shfl_xor(sum_all[r], o, 32);
      spos[r]    += __shfl_xor(spos[r],   o, 32);
      cpos[r]    += __shfl_xor(cpos[r],   o, 32);
      mopp[r]    = fmaxf(mopp[r], __shfl_xor(mopp[r], o, 32));
      sopp[r]    += __shfl_xor(sopp[r],   o, 32);
      firsts[r]  = fmaxf(firsts[r], __shfl_xor(firsts[r], o, 32));
    }
  }

  if (ln == 0) {
#pragma unroll
    for (int r = 0; r < 8; ++r) {
      int row = r + 8 * half;
      comb[w][row][0] = mx_all[r];
      comb[w][row][1] = sum_all[r];
      comb[w][row][2] = spos[r];
      comb[w][row][3] = cpos[r];
      comb[w][row][4] = mopp[r];
      comb[w][row][5] = sopp[r];
      comb[w][row][6] = firsts[r];
    }
  }
  __syncthreads();

  // --- combine the waves' column ranges, emit per-row loss terms
  if (tid < 16) {
    float mxa = NEGBIG, sa = 0.0f, sp = 0.0f, cp = 0.0f;
    float mo  = NEGBIG, so = 0.0f, fs = NEGBIG;
#pragma unroll
    for (int ww = 0; ww < NWAVES; ++ww) {
      mxa = fmaxf(mxa, comb[ww][tid][0]);
      sa += comb[ww][tid][1];
      sp += comb[ww][tid][2];
      cp += comb[ww][tid][3];
      mo  = fmaxf(mo, comb[ww][tid][4]);
      so += comb[ww][tid][5];
      fs  = fmaxf(fs, comb[ww][tid][6]);
    }
    const int i = row_base + tid;
    const float wgt = weight[labels[i]];
    // log sum_{j!=i} exp(s - mxa) = log(sa) + C - mxa
    const float Lall = __logf(sa) + CMAX - mxa;
    // -mean_log_prob_pos = mxa + Lall - sp/cp   (cp >= 1: column i+B always positive)
    rowloss[i] = wgt * (mxa + Lall - sp / cp);
    // reference: -(first - log(denom)),  first = s[i,i+B]-mxa,
    // denom = sum_opp exp(s - mo)  =>  log(denom) = log(so) + C - mo
    rowelim[i] = wgt * (__logf(so) + CMAX - mo - (fs - mxa));
  }
}

// ---------------------------------------------------------------------------
// Kernel 3: deterministic single-block reduction + finalize.
// ---------------------------------------------------------------------------
__global__ void supcon_finalize(const float* __restrict__ rowloss,
                                const float* __restrict__ rowelim,
                                const int*   __restrict__ labels,
                                float* __restrict__ out) {
  __shared__ float sl[256], se[256];
  __shared__ int   sc[256];
  const int t = threadIdx.x;
  float l = 0.0f, e = 0.0f;
  int c = 0;
  for (int i = t; i < BROWS; i += 256) {
    l += rowloss[i];
    e += rowelim[i];
    c += labels[i];
  }
  sl[t] = l; se[t] = e; sc[t] = c;
  __syncthreads();
  for (int o = 128; o > 0; o >>= 1) {
    if (t < o) { sl[t] += sl[t + o]; se[t] += se[t + o]; sc[t] += sc[t + o]; }
    __syncthreads();
  }
  if (t == 0) {
    bool both = (sc[0] > 0) && (sc[0] < BROWS);
    float loss = sl[0];
    float elim = both ? se[0] : loss;
    out[0] = loss / (float)BROWS + elim / (float)(2 * BROWS);
  }
}

// ---------------------------------------------------------------------------
// Workspace layout (bytes):
//   [0,      24576)  rowloss[6144] f32
//   [24576,  49152)  rowelim[6144] f32
//   [49152,  +3.0MB) contrast bf16 hi [12288*128]
//   [next,   +3.0MB) contrast bf16 lo [12288*128]
// Total ~6.34 MB.
// ---------------------------------------------------------------------------
extern "C" void kernel_launch(void* const* d_in, const int* in_sizes, int n_in,
                              void* d_out, int out_size, void* d_ws, size_t ws_size,
                              hipStream_t stream) {
  (void)in_sizes; (void)n_in; (void)out_size; (void)ws_size;
  const float* feat   = (const float*)d_in[0];
  const int*   labels = (const int*)d_in[1];
  const float* weight = (const float*)d_in[2];
  float* out = (float*)d_out;

  char* ws = (char*)d_ws;
  float* rowloss = (float*)(ws);
  float* rowelim = (float*)(ws + 24576);
  unsigned short* hi = (unsigned short*)(ws + 49152);
  unsigned short* lo = (unsigned short*)(ws + 49152 + (size_t)NCOL * DIM * 2);

  supcon_prep<<<(NCOL * DIM) / 256, 256, 0, stream>>>(feat, hi, lo);
  supcon_main<<<BROWS / 16, NWAVES * 32, 0, stream>>>(hi, lo, labels, weight, rowloss, rowelim);
  supcon_finalize<<<1, 256, 0, stream>>>(rowloss, rowelim, labels, out);
}
